// ForcedCollapseAttention_63118839382525
// MI455X (gfx1250) — matrix operand order repaired and checked by reading, not compiled, not run on hardware
//
#include <hip/hip_runtime.h>
#include <hip/hip_bf16.h>

// ---------------------------------------------------------------------------
// CDNA5 (gfx1250) transformer layer, all GEMMs via v_wmma_f32_16x16x32_f16.
// wave32; WMMA fragment layouts per CDNA5 ISA 7.12.2:
//   A (16x32 f16): lane<16 holds K0..7,K16..23 ; lane>=16 holds K8..15,K24..31
//   B (32x16 f16): lane<16 holds K0..7,K16..23       ; lane>=16 holds K8..15,K24..31
//   C/D (16x16 f32): vgpr r, lane l -> M = (l<16 ? r : r+8), N = l&15
// ---------------------------------------------------------------------------

typedef _Float16 h16;
typedef h16   v16h __attribute__((ext_vector_type(16)));
typedef h16   v8h  __attribute__((ext_vector_type(8)));
typedef float v8f  __attribute__((ext_vector_type(8)));

static __device__ __forceinline__ v8f wmma_f16(v16h a, v16h b, v8f c) {
  return __builtin_amdgcn_wmma_f32_16x16x32_f16(false, a, false, b, (short)0, c, false, false);
}

// A-fragment: rowp = start of this lane's row (row-major, K contiguous)
static __device__ __forceinline__ v16h load_fragA(const h16* rowp, int lhi) {
  const char* p = (const char*)rowp + lhi * 16;
  v16h a;
  ((v8h*)&a)[0] = *(const v8h*)(p);        // K 0..7   (or 8..15)
  ((v8h*)&a)[1] = *(const v8h*)(p + 32);   // K 16..23 (or 24..31)
  return a;
}
// B-fragment: rowp = start of this lane's row of W^T (i.e. column of B)
static __device__ __forceinline__ v16h load_fragB(const h16* rowp, int lhi) {
  const char* p = (const char*)rowp + lhi * 16;
  v16h b;
  ((v8h*)&b)[0] = *(const v8h*)(p);        // K 0..7   (or 8..15)
  ((v8h*)&b)[1] = *(const v8h*)(p + 32);   // K 16..23 (or 24..31)
  return b;
}

enum { EPI_HEADS = 0, EPI_HEADS_T = 1, EPI_RELU = 2, EPI_RESID_F32 = 3, EPI_ATTNV = 4 };

// ---------------------------------------------------------------------------
// Generic tiled WMMA GEMM: C[M,N] = A[M,K] * W^T  (W stored [N,K] row-major)
// 256 threads = 8 waves, tile 128x128x32, wave computes 32x64.
// Double-buffered LDS (ping-pong), one barrier per k-step; global prefetch
// two tiles ahead (global_prefetch_b8).
// ---------------------------------------------------------------------------
template <int EPI>
__global__ __launch_bounds__(256) void gemm_f16(
    const h16* __restrict__ A, const h16* __restrict__ W,
    const float* __restrict__ bias, const float* __restrict__ resid,
    float* __restrict__ outF, h16* __restrict__ outB,
    int M, int N, int K, long long sAb, long long sWb)
{
  __shared__ __align__(16) h16 As[2][128 * 40];   // rows padded to 80B: bank-clean
  __shared__ __align__(16) h16 Ws[2][128 * 40];

  const int t = threadIdx.x, lane = t & 31, w = t >> 5;
  const int wm = w & 3, wn = w >> 2;              // 4 x 2 wave grid
  const int bm = blockIdx.y * 128, bn = blockIdx.x * 128;
  const long long z = blockIdx.z;
  const h16* Ab = A + z * sAb + (long long)bm * K;
  const h16* Wb = W + z * sWb + (long long)bn * K;
  const int l15 = lane & 15, lhi = lane >> 4;

  // per-thread tile-load coordinates: chunks t and t+256 of 512 x 16B
  const int r0 = t >> 2, cc0 = (t & 3) * 8;       // rows r0 and r0+64, same col
  const long long offA0 = (long long)r0 * K + cc0;
  const long long offA1 = (long long)(r0 + 64) * K + cc0;
  const int ldsOff0 = r0 * 40 + cc0, ldsOff1 = (r0 + 64) * 40 + cc0;

  const v8f vzero = {0.f,0.f,0.f,0.f,0.f,0.f,0.f,0.f};
  v8f acc[2][4];
#pragma unroll
  for (int i = 0; i < 2; ++i)
#pragma unroll
    for (int j = 0; j < 4; ++j) acc[i][j] = vzero;

  const int nk = K >> 5;

  // prologue: tile 0 -> buffer 0
  {
    v8h a0 = *(const v8h*)(Ab + offA0);
    v8h a1 = *(const v8h*)(Ab + offA1);
    v8h w0 = *(const v8h*)(Wb + offA0);
    v8h w1 = *(const v8h*)(Wb + offA1);
    *(v8h*)&As[0][ldsOff0] = a0;  *(v8h*)&As[0][ldsOff1] = a1;
    *(v8h*)&Ws[0][ldsOff0] = w0;  *(v8h*)&Ws[0][ldsOff1] = w1;
  }
  __syncthreads();

  for (int kt = 0; kt < nk; ++kt) {
    const int cur = kt & 1;
    const bool more = (kt + 1) < nk;

    // stage next tile into registers (overlaps with WMMAs below)
    v8h a0, a1, w0, w1;
    if (more) {
      const int kn = (kt + 1) << 5;
      a0 = *(const v8h*)(Ab + kn + offA0);
      a1 = *(const v8h*)(Ab + kn + offA1);
      w0 = *(const v8h*)(Wb + kn + offA0);
      w1 = *(const v8h*)(Wb + kn + offA1);
      if (kt + 2 < nk) {                          // warm GL2/WGP$ two tiles out
        const int kp = (kt + 2) << 5;
        __builtin_prefetch(Ab + kp + offA0, 0, 1);
        __builtin_prefetch(Wb + kp + offA0, 0, 1);
      }
    }

    // compute on current buffer
    v16h af[2], bfr[4];
#pragma unroll
    for (int i = 0; i < 2; ++i) af[i]  = load_fragA(&As[cur][(wm * 32 + i * 16 + l15) * 40], lhi);
#pragma unroll
    for (int j = 0; j < 4; ++j) bfr[j] = load_fragB(&Ws[cur][(wn * 64 + j * 16 + l15) * 40], lhi);
#pragma unroll
    for (int i = 0; i < 2; ++i)
#pragma unroll
      for (int j = 0; j < 4; ++j)
        acc[i][j] = wmma_f16(af[i], bfr[j], acc[i][j]);

    // commit staged tile to the other buffer
    if (more) {
      const int nxt = cur ^ 1;
      *(v8h*)&As[nxt][ldsOff0] = a0;  *(v8h*)&As[nxt][ldsOff1] = a1;
      *(v8h*)&Ws[nxt][ldsOff0] = w0;  *(v8h*)&Ws[nxt][ldsOff1] = w1;
    }
    __syncthreads();
  }

  // epilogue
#pragma unroll
  for (int i = 0; i < 2; ++i)
#pragma unroll
    for (int j = 0; j < 4; ++j) {
      const int gn = bn + wn * 64 + j * 16 + l15;
#pragma unroll
      for (int r = 0; r < 8; ++r) {
        const int gm = bm + wm * 32 + i * 16 + (lhi ? r + 8 : r);
        float v = acc[i][j][r];
        if constexpr (EPI == EPI_HEADS) {           // q/k: [B,H,L,HD] f16
          v += bias[gn];
          const int bb = gm >> 9, l = gm & 511, hh = gn >> 8, hd = gn & 255;
          outB[(((long long)(bb * 4 + hh) * 512 + l) << 8) + hd] = (h16)v;
        } else if constexpr (EPI == EPI_HEADS_T) {  // v: [B,H,HD,L] f16
          v += bias[gn];
          const int bb = gm >> 9, l = gm & 511, hh = gn >> 8, hd = gn & 255;
          outB[(((long long)(bb * 4 + hh) * 256 + hd) << 9) + l] = (h16)v;
        } else if constexpr (EPI == EPI_RELU) {
          v = fmaxf(v + bias[gn], 0.f);
          outB[(long long)gm * N + gn] = (h16)v;
        } else if constexpr (EPI == EPI_RESID_F32) {
          const long long o = (long long)gm * N + gn;
          outF[o] = v + bias[gn] + resid[o];
        } else if constexpr (EPI == EPI_ATTNV) {    // O: [B*L, D] f16, batched
          const int bb = (int)z >> 2, hh = (int)z & 3;
          outB[(long long)(bb * 512 + gm) * 1024 + hh * 256 + gn] = (h16)v;
        }
      }
    }
}

// ---------------------------------------------------------------------------
// Weight convert + transpose: w[K,N] f32 -> wt[N,K] f16
// ---------------------------------------------------------------------------
__global__ void cvt_w_t(const float* __restrict__ w, h16* __restrict__ wt, int K, int N) {
  const long long idx = (long long)blockIdx.x * 256 + threadIdx.x;
  if (idx < (long long)K * N) {
    const int kk = (int)(idx / N), n = (int)(idx % N);
    wt[(long long)n * K + kk] = (h16)w[idx];
  }
}

// ---------------------------------------------------------------------------
// LayerNorm over D=1024, one block per row. g/b optional. f32 in, f16 out.
// ---------------------------------------------------------------------------
__global__ __launch_bounds__(256) void ln_f32_to_f16(
    const float* __restrict__ x, const float* __restrict__ g,
    const float* __restrict__ bta, h16* __restrict__ out)
{
  __shared__ float rs[256], rq[256];
  const int t = threadIdx.x;
  const long long row = blockIdx.x;
  const float* xr = x + row * 1024;
  float s = 0.f, q = 0.f;
  for (int i = t; i < 1024; i += 256) { const float v = xr[i]; s += v; q += v * v; }
  rs[t] = s; rq[t] = q; __syncthreads();
  for (int o = 128; o > 0; o >>= 1) {
    if (t < o) { rs[t] += rs[t + o]; rq[t] += rq[t + o]; }
    __syncthreads();
  }
  const float mean = rs[0] * (1.f / 1024.f);
  const float var  = rq[0] * (1.f / 1024.f) - mean * mean;
  const float rstd = rsqrtf(var + 1e-5f);
  for (int i = t; i < 1024; i += 256) {
    float y = (xr[i] - mean) * rstd;
    if (g) y = y * g[i] + bta[i];
    out[row * 1024 + i] = (h16)y;
  }
}

// ---------------------------------------------------------------------------
// fw bias vector: std(ddof=1) gate, conditional adjust, * fs/temp * 3.0
// ---------------------------------------------------------------------------
__global__ void fw_bias_kernel(const float* __restrict__ fw, const float* __restrict__ temp,
                               const float* __restrict__ fs, float* __restrict__ fwB)
{
  __shared__ float rs[256], rq[256];
  const int t = threadIdx.x;
  const float a = fw[t], b = fw[t + 256];
  rs[t] = a + b; rq[t] = a * a + b * b;
  __syncthreads();
  for (int o = 128; o > 0; o >>= 1) {
    if (t < o) { rs[t] += rs[t + o]; rq[t] += rq[t + o]; }
    __syncthreads();
  }
  const float sum = rs[0], sq = rq[0];
  const float mean = sum * (1.f / 512.f);
  const float var  = (sq - sum * mean) * (1.f / 511.f);
  const bool  adj  = sqrtf(var) < 0.5f;
  const float scale = fs[0] / temp[0] * 3.0f;
  for (int i = t; i < 512; i += 256) {
    float v = fw[i];
    if (adj) v += (i < 5) ? 2.0f : -0.5f;
    fwB[i] = v * scale;
  }
}

// ---------------------------------------------------------------------------
// Fused attention scores + softmax per (qtile, h, b).
// 16 q-rows x 512 k-cols. 8 waves * 4 n-tiles. f16 attn out + f32 head-0 out.
// ---------------------------------------------------------------------------
__global__ __launch_bounds__(256) void attn_scores_softmax(
    const h16* __restrict__ q, const h16* __restrict__ k,
    const float* __restrict__ fwB, h16* __restrict__ attnB,
    float* __restrict__ attnF)
{
  __shared__ __align__(16) h16  q_s[16 * 264];   // pad 264: 528B row stride
  __shared__ float sc[16 * 520];                 // scores, pad 520
  __shared__ float red[16][17];
  __shared__ float rowstat[16];

  const int qt = blockIdx.x, h = blockIdx.y, b = blockIdx.z;
  const int t = threadIdx.x, lane = t & 31, w = t >> 5;
  const long long headoff = (long long)(b * 4 + h) * 512 * 256;
  const h16* qh = q + headoff + (long long)qt * 16 * 256;
  const h16* kh = k + headoff;

  for (int c = t; c < 512; c += 256) {           // 16 rows x 512B
    const int row = c >> 5, cc = c & 31;
    *(v8h*)&q_s[row * 264 + cc * 8] = *(const v8h*)(qh + row * 256 + cc * 8);
  }
  __syncthreads();

  const int l15 = lane & 15, lhi = lane >> 4;
  const v8f vzero = {0.f,0.f,0.f,0.f,0.f,0.f,0.f,0.f};
  v8f acc[4] = {vzero, vzero, vzero, vzero};
  const int colbase = w * 64;

#pragma unroll
  for (int k0 = 0; k0 < 256; k0 += 32) {
    const v16h af = load_fragA(&q_s[l15 * 264 + k0], lhi);
#pragma unroll
    for (int j = 0; j < 4; ++j) {
      const h16* krow = kh + (long long)(colbase + j * 16 + l15) * 256 + k0;
      acc[j] = wmma_f16(af, load_fragB(krow, lhi), acc[j]);
    }
  }

  const float scale = 0.0625f;                   // 1/sqrt(256)
#pragma unroll
  for (int j = 0; j < 4; ++j) {
    const int col = colbase + j * 16 + l15;
#pragma unroll
    for (int r = 0; r < 8; ++r) {
      const int row = lhi ? r + 8 : r;
      float v = acc[j][r] * scale;
      if (qt == 0 && row == 0) v += fwB[col];    // bias on query row 0, all heads
      sc[row * 520 + col] = v;
    }
  }
  __syncthreads();

  // softmax over 512 cols: 16 threads per row, stride-16 columns
  const int srow = t >> 4, c0 = t & 15;
  float m = -1e30f;
  for (int i = 0; i < 32; ++i) m = fmaxf(m, sc[srow * 520 + c0 + 16 * i]);
  red[srow][c0] = m; __syncthreads();
  if (c0 == 0) {
    float mm = red[srow][0];
    for (int i = 1; i < 16; ++i) mm = fmaxf(mm, red[srow][i]);
    rowstat[srow] = mm;
  }
  __syncthreads();
  const float rmax = rowstat[srow];
  float ssum = 0.f;
  for (int i = 0; i < 32; ++i) {
    const float e = expf(sc[srow * 520 + c0 + 16 * i] - rmax);
    sc[srow * 520 + c0 + 16 * i] = e; ssum += e;
  }
  red[srow][c0] = ssum; __syncthreads();
  if (c0 == 0) {
    float s2 = 0.f;
    for (int i = 0; i < 16; ++i) s2 += red[srow][i];
    rowstat[srow] = s2;
  }
  __syncthreads();
  const float inv = 1.0f / rowstat[srow];
  const int qg = qt * 16 + srow;
  h16*   aB = attnB + ((long long)(b * 4 + h) * 512 + qg) * 512;
  float* aF = attnF + ((long long)b * 512 + qg) * 512;
  for (int i = 0; i < 32; ++i) {
    const int col = c0 + 16 * i;
    const float p = sc[srow * 520 + col] * inv;
    aB[col] = (h16)p;
    if (h == 0) aF[col] = p;
  }
}

// ---------------------------------------------------------------------------
// Per-batch entropy collapse on attn[b, head0, q0, :]
// ---------------------------------------------------------------------------
__global__ void entropy_collapse(float* __restrict__ attnF, h16* __restrict__ attnB) {
  __shared__ float rs[256];
  const int b = blockIdx.x, t = threadIdx.x;
  float* row = attnF + (long long)b * 512 * 512;        // q = 0 row of head 0
  float e = 0.f;
  for (int i = t; i < 512; i += 256) { const float p = row[i]; e -= p * logf(p + 1e-8f); }
  rs[t] = e; __syncthreads();
  for (int o = 128; o > 0; o >>= 1) {
    if (t < o) rs[t] += rs[t + o];
    __syncthreads();
  }
  const float ent = rs[0];
  if (ent * (1.0f / logf(512.f)) > 0.9f) {
    h16* brow = attnB + (long long)(b * 4) * 512 * 512; // h=0, q=0
    for (int i = t; i < 512; i += 256) {
      const float nv = (i < 3) ? (0.7f / 3.0f) : (0.3f / 509.0f);
      row[i] = nv; brow[i] = (h16)nv;
    }
  }
}

// ---------------------------------------------------------------------------
extern "C" void kernel_launch(void* const* d_in, const int* in_sizes, int n_in,
                              void* d_out, int out_size, void* d_ws, size_t ws_size,
                              hipStream_t stream) {
  (void)in_sizes; (void)n_in; (void)out_size; (void)ws_size;
  const float* s    = (const float*)d_in[0];
  const float* zin  = (const float*)d_in[1];
  const float* wq   = (const float*)d_in[2];  const float* bq = (const float*)d_in[3];
  const float* wk   = (const float*)d_in[4];  const float* bk = (const float*)d_in[5];
  const float* wv   = (const float*)d_in[6];  const float* bv = (const float*)d_in[7];
  const float* wo   = (const float*)d_in[8];  const float* bo = (const float*)d_in[9];
  const float* fw   = (const float*)d_in[10];
  const float* temp = (const float*)d_in[11];
  const float* fs   = (const float*)d_in[12];
  const float* lng  = (const float*)d_in[13]; const float* lnb = (const float*)d_in[14];
  const float* w1   = (const float*)d_in[15]; const float* b1  = (const float*)d_in[16];
  const float* w2   = (const float*)d_in[17]; const float* b2  = (const float*)d_in[18];

  float* out_s    = (float*)d_out;           // 8192x1024
  float* out_z    = out_s + 8388608;         // 16x512x256
  float* out_attn = out_z + 2097152;         // 16x512x512 (head 0)

  // workspace layout (256B aligned), with aliasing of dead regions
  char* p = (char*)d_ws;
  auto alloc = [&](size_t bytes) { char* r = p; p += (bytes + 255) & ~(size_t)255; return r; };
  h16*  s_ln  = (h16*)alloc(8192ull * 1024 * 2);      // aliased later by h_ln
  h16*  wq_t  = (h16*)alloc(1024ull * 1024 * 2);
  h16*  wk_t  = (h16*)alloc(1024ull * 1024 * 2);
  h16*  wv_t  = (h16*)alloc(1024ull * 1024 * 2);
  h16*  wo_t  = (h16*)alloc(1024ull * 1024 * 2);
  h16*  w1_t  = (h16*)alloc(4096ull * 1024 * 2);
  h16*  w2_t  = (h16*)alloc(1024ull * 4096 * 2);
  h16*  qbuf  = (h16*)alloc(64ull * 512 * 256 * 2);   // [B,H,L,HD]
  h16*  kbuf  = (h16*)alloc(64ull * 512 * 256 * 2);
  h16*  vtbuf = (h16*)alloc(64ull * 256 * 512 * 2);   // [B,H,HD,L]
  h16*  attnB = (h16*)alloc(64ull * 512 * 512 * 2);   // [B,H,L,L]
  h16*  Obuf  = (h16*)alloc(8192ull * 1024 * 2);      // [B*L, D]
  float* s1   = (float*)alloc(8192ull * 1024 * 4);
  float* fwB  = (float*)alloc(512 * 4);
  h16*  h_ln  = s_ln;                                  // alias: s_ln dead after QKV
  h16*  ffn1  = qbuf;                                  // alias: q..attn dead after wo GEMM

  // 1) weight convert + transpose (f32 [K,N] -> f16 [N,K])
  cvt_w_t<<<4096, 256, 0, stream>>>(wq, wq_t, 1024, 1024);
  cvt_w_t<<<4096, 256, 0, stream>>>(wk, wk_t, 1024, 1024);
  cvt_w_t<<<4096, 256, 0, stream>>>(wv, wv_t, 1024, 1024);
  cvt_w_t<<<4096, 256, 0, stream>>>(wo, wo_t, 1024, 1024);
  cvt_w_t<<<16384, 256, 0, stream>>>(w1, w1_t, 1024, 4096);
  cvt_w_t<<<16384, 256, 0, stream>>>(w2, w2_t, 4096, 1024);

  // 2) LN1 (no affine) and fw bias vector
  ln_f32_to_f16<<<8192, 256, 0, stream>>>(s, nullptr, nullptr, s_ln);
  fw_bias_kernel<<<1, 256, 0, stream>>>(fw, temp, fs, fwB);

  // 3) QKV projections
  gemm_f16<EPI_HEADS><<<dim3(8, 64, 1), 256, 0, stream>>>(
      s_ln, wq_t, bq, nullptr, nullptr, qbuf, 8192, 1024, 1024, 0, 0);
  gemm_f16<EPI_HEADS><<<dim3(8, 64, 1), 256, 0, stream>>>(
      s_ln, wk_t, bk, nullptr, nullptr, kbuf, 8192, 1024, 1024, 0, 0);
  gemm_f16<EPI_HEADS_T><<<dim3(8, 64, 1), 256, 0, stream>>>(
      s_ln, wv_t, bv, nullptr, nullptr, vtbuf, 8192, 1024, 1024, 0, 0);

  // 4) scores + bias + softmax, then entropy collapse
  attn_scores_softmax<<<dim3(32, 4, 16), 256, 0, stream>>>(qbuf, kbuf, fwB, attnB, out_attn);
  entropy_collapse<<<16, 256, 0, stream>>>(out_attn, attnB);

  // 5) attn @ v  (batched over 64 heads) -> O [B*L, D]
  gemm_f16<EPI_ATTNV><<<dim3(2, 4, 64), 256, 0, stream>>>(
      attnB, vtbuf, nullptr, nullptr, nullptr, Obuf,
      512, 256, 512, 512LL * 512, 256LL * 512);

  // 6) output projection + residual -> s1 (f32)
  gemm_f16<EPI_RESID_F32><<<dim3(8, 64, 1), 256, 0, stream>>>(
      Obuf, wo_t, bo, s, s1, nullptr, 8192, 1024, 1024, 0, 0);

  // 7) LN2 (affine) -> h
  ln_f32_to_f16<<<8192, 256, 0, stream>>>(s1, lng, lnb, h_ln);

  // 8) FFN
  gemm_f16<EPI_RELU><<<dim3(32, 64, 1), 256, 0, stream>>>(
      h_ln, w1_t, b1, nullptr, nullptr, ffn1, 8192, 4096, 1024, 0, 0);
  gemm_f16<EPI_RESID_F32><<<dim3(8, 64, 1), 256, 0, stream>>>(
      ffn1, w2_t, b2, s1, out_s, nullptr, 8192, 1024, 4096, 0, 0);

  // 9) z passthrough
  hipMemcpyAsync(out_z, zin, 2097152ull * 4, hipMemcpyDeviceToDevice, stream);
}
